// SwinTransformerBlock_8701603742150
// MI455X (gfx1250) — matrix operand order repaired and checked by reading, not compiled, not run on hardware
//
#include <hip/hip_runtime.h>
#include <hip/hip_bf16.h>
#include <math.h>

// ---------------- problem constants ----------------
#define C_DIM     96
#define HEADS     3
#define HD        32
#define WSZ       7
#define SHIFT3    3
#define RS        28
#define RH        56
#define RW        56
#define NTOK      343           // 7^3
#define NPAD      352           // 22 * 16
#define NWIN      256           // 4*8*8
#define LTOK      87808         // 256*343
#define MTILES    5488          // LTOK / 16
#define MLPH      384

typedef __attribute__((ext_vector_type(16))) _Float16 v16h;
typedef __attribute__((ext_vector_type(8)))  _Float16 v8h;
typedef __attribute__((ext_vector_type(8)))  float    v8f;

static __device__ __forceinline__ v8f wmma16(v16h a, v16h b, v8f c) {
  // D = A(16x32 f16) * B(32x16 f16) + C(16x16 f32)
  return __builtin_amdgcn_wmma_f32_16x16x32_f16(false, a, false, b, (short)0, c,
                                                false, false);
}

// A fragment: lane<16 -> M=lane, k0..k0+15;
// lane>=16 -> M=lane-16, k0+16..k0+31.
// base must point at (row, k0 + hi*16); 16-byte aligned.
static __device__ __forceinline__ v16h load_afrag(const _Float16* base) {
  v16h a;
  *(v8h*)&a       = *(const v8h*)base;
  *((v8h*)&a + 1) = *(const v8h*)(base + 8);
  return a;
}

// map (window b_, token n) -> flat token index in the original (rolled-back) layout.
// Both the forward gather (roll -3 then partition) and the reverse scatter
// (window reverse then roll +3) reduce to the same +SHIFT mod mapping.
static __device__ __forceinline__ int win_to_flat(int b_, int n) {
  int sb = b_ >> 6, hb = (b_ >> 3) & 7, wb = b_ & 7;
  int zi = n / 49, yi = (n / 7) % 7, xi = n % 7;
  int z = sb * WSZ + zi, y = hb * WSZ + yi, x = wb * WSZ + xi;
  int zp = (z + SHIFT3) % RS;
  int yp = (y + SHIFT3) % RH;
  int xp = (x + SHIFT3) % RW;
  return (zp * RH + yp) * RW + xp;
}

// ---------------- kernel 0: weight conversion + scale ----------------
__global__ void prep_kernel(const float* qkv_w, const float* proj_w,
                            const float* fc1_w, const float* fc2_w,
                            const float* logit_scale,
                            _Float16* qkvh, _Float16* projh,
                            _Float16* fc1h, _Float16* fc2h, float* scale) {
  int i = blockIdx.x * 256 + threadIdx.x;
  const int n0 = 3 * C_DIM * C_DIM;        // 27648
  const int n1 = n0 + C_DIM * C_DIM;       // 36864
  const int n2 = n1 + MLPH * C_DIM;        // 73728
  const int n3 = n2 + C_DIM * MLPH;        // 110592
  if (i < n0)       qkvh[i]      = (_Float16)qkv_w[i];
  else if (i < n1)  projh[i-n0]  = (_Float16)proj_w[i-n0];
  else if (i < n2)  fc1h[i-n1]   = (_Float16)fc1_w[i-n1];
  else if (i < n3)  fc2h[i-n2]   = (_Float16)fc2_w[i-n2];
  if (i < HEADS) scale[i] = __expf(fminf(logit_scale[i], logf(100.0f)));
}

// ---------------- kernel 1: roll+partition fused QKV GEMM ----------------
// out: q/k/v f16 [win][head][NPAD][HD]  (pad rows 343..351 left unwritten)
__global__ void qkv_kernel(const float* __restrict__ x,
                           const _Float16* __restrict__ wq,
                           const float* __restrict__ qkv_b,
                           _Float16* __restrict__ qh,
                           _Float16* __restrict__ kh,
                           _Float16* __restrict__ vh) {
  __shared__ __attribute__((aligned(64))) _Float16 sA[16 * C_DIM];
  const int tile = blockIdx.x;
  const int tid  = threadIdx.x;

  for (int i = tid; i < 16 * C_DIM; i += 32) {
    int r = i / C_DIM, c = i - r * C_DIM;
    int g = tile * 16 + r;
    int b_ = g / NTOK, n = g - b_ * NTOK;
    int l = win_to_flat(b_, n);
    sA[i] = (_Float16)x[(size_t)l * C_DIM + c];
  }
  __syncthreads();

  const int mr = tid & 15;
  const int hi = tid >> 4;

  v16h afr[3];
  for (int kc = 0; kc < 3; ++kc)
    afr[kc] = load_afrag(sA + mr * C_DIM + kc * 32 + hi * 16);

  for (int nt = 0; nt < 18; ++nt) {
    v8f acc = {};
    for (int kc = 0; kc < 3; ++kc) {
      v16h b = *(const v16h*)(wq + (nt * 16 + mr) * C_DIM + kc * 32 + hi * 16);
      acc = wmma16(afr[kc], b, acc);
    }
    int col = nt * 16 + mr;
    float bias = qkv_b[col];
    int which = col / C_DIM;
    int rem = col - which * C_DIM;
    int head = rem >> 5, d = rem & 31;
    _Float16* dst = (which == 0) ? qh : (which == 1) ? kh : vh;
    for (int r = 0; r < 8; ++r) {
      int g = tile * 16 + r + hi * 8;
      int b_ = g / NTOK, n = g - b_ * NTOK;
      dst[(((size_t)(b_ * 3 + head)) * NPAD + n) * HD + d] =
          (_Float16)(acc[r] + bias);
    }
  }
}

// ---------------- kernel 2: windowed cosine attention ----------------
// grid (22 q-tiles, 3 heads, 256 windows), 32 threads (1 wave).
__global__ void attn_kernel(const _Float16* __restrict__ qh,
                            const _Float16* __restrict__ kh,
                            const _Float16* __restrict__ vh,
                            const float* __restrict__ scale,
                            const float* __restrict__ relTable,
                            const float* __restrict__ mask,
                            _Float16* __restrict__ attnout) {
  __shared__ __attribute__((aligned(64))) _Float16 sK [NPAD * HD];   // [key][d]
  __shared__ __attribute__((aligned(64))) _Float16 sVt[HD * NPAD];   // [d][key]
  __shared__ __attribute__((aligned(64))) _Float16 sSP[16 * NPAD];   // logits -> exp (in place)
  __shared__ float sSum[16];

  const int tid = threadIdx.x;
  const int t = blockIdx.x, h = blockIdx.y, w = blockIdx.z;
  const int mr = tid & 15, hi = tid >> 4;

  const _Float16* gK = kh + ((size_t)(w * 3 + h)) * NPAD * HD;
  const _Float16* gV = vh + ((size_t)(w * 3 + h)) * NPAD * HD;
  const _Float16* gQ = qh + ((size_t)(w * 3 + h)) * NPAD * HD;

  // load K (zero the pad rows) and V transposed (zero the pad rows)
  for (int i = tid; i < NPAD * (HD / 2); i += 32) {
    int row = i >> 4;
    unsigned val = (row < NTOK) ? ((const unsigned*)gK)[i] : 0u;
    ((unsigned*)sK)[i] = val;
  }
  for (int i = tid; i < NPAD * HD; i += 32) {
    int row = i >> 5, d = i & 31;
    _Float16 val = (row < NTOK) ? gV[i] : (_Float16)0.0f;
    sVt[d * NPAD + row] = val;
  }
  __syncthreads();

  // normalize K rows: kn = k / max(||k||, 1e-12)
  for (int row = tid; row < NPAD; row += 32) {
    float ss = 0.0f;
    for (int d = 0; d < HD; ++d) { float v = (float)sK[row * HD + d]; ss += v * v; }
    float inv = 1.0f / fmaxf(sqrtf(ss), 1e-12f);
    for (int d = 0; d < HD; ++d)
      sK[row * HD + d] = (_Float16)((float)sK[row * HD + d] * inv);
  }

  // build normalized+scaled q fragment (16x32) in registers
  const int nq = t * 16 + mr;
  float qv[16];
  float ss = 0.0f;
  if (nq < NTOK) {
    for (int j = 0; j < 8; ++j) {
      qv[j]     = (float)gQ[nq * HD + hi * 16 + j];
      qv[8 + j] = (float)gQ[nq * HD + hi * 16 + 8 + j];
      ss += qv[j] * qv[j] + qv[8 + j] * qv[8 + j];
    }
  } else {
    for (int j = 0; j < 16; ++j) qv[j] = 0.0f;
  }
  ss += __shfl_xor(ss, 16);
  float sc = scale[h] / fmaxf(sqrtf(ss), 1e-12f);
  v16h aq;
  for (int j = 0; j < 16; ++j) aq[j] = (_Float16)(qv[j] * sc);
  __syncthreads();

  // S = qn @ kn^T : 22 column tiles
  for (int j = 0; j < 22; ++j) {
    v16h b = *(const v16h*)(sK + (j * 16 + mr) * HD + hi * 16);
    v8f acc = {};
    acc = wmma16(aq, b, acc);
    for (int r = 0; r < 8; ++r)
      sSP[(r + hi * 8) * NPAD + j * 16 + mr] = (_Float16)acc[r];
  }
  __syncthreads();

  // softmax with relative position bias + mask; two lanes per row
  {
    const int row = mr;
    const int n_g = t * 16 + row;
    int qz = 0, qy = 0, qx = 0;
    if (n_g < NTOK) { qz = n_g / 49; qy = (n_g / 7) % 7; qx = n_g % 7; }
    const int m0 = hi * (NPAD / 2);
    float mx = -1e30f;
    for (int m = m0; m < m0 + NPAD / 2; ++m) {
      float val;
      if (n_g < NTOK && m < NTOK) {
        int mz = m / 49, my = (m / 7) % 7, mxx = m % 7;
        int idx = (qz - mz + 6) * 20 + (qy - my + 6) * 13 + (qx - mxx + 6);
        val = (float)sSP[row * NPAD + m] + relTable[idx * HEADS + h] +
              mask[((size_t)w * NTOK + n_g) * NTOK + m];
      } else {
        val = -60000.0f;           // f16-safe -inf substitute
      }
      sSP[row * NPAD + m] = (_Float16)val;
      mx = fmaxf(mx, val);
    }
    mx = fmaxf(mx, __shfl_xor(mx, 16));
    float sum = 0.0f;
    for (int m = m0; m < m0 + NPAD / 2; ++m) {
      float p = __expf((float)sSP[row * NPAD + m] - mx);
      sSP[row * NPAD + m] = (_Float16)p;
      sum += p;
    }
    sum += __shfl_xor(sum, 16);
    if (hi == 0) sSum[row] = 1.0f / sum;
  }
  __syncthreads();

  // O = P @ V : 2 output column tiles, 11 K-chunks of 32 each
  for (int c = 0; c < 2; ++c) {
    v8f acc = {};
    for (int kc = 0; kc < 11; ++kc) {
      v16h a = load_afrag(sSP + mr * NPAD + kc * 32 + hi * 16);
      v16h b = *(const v16h*)(sVt + (c * 16 + mr) * NPAD + kc * 32 + hi * 16);
      acc = wmma16(a, b, acc);
    }
    for (int r = 0; r < 8; ++r) {
      int rl = r + hi * 8;
      int n = t * 16 + rl;
      if (n < NTOK) {
        float o = acc[r] * sSum[rl];
        attnout[((size_t)(w * NTOK + n)) * C_DIM + h * HD + c * 16 + mr] =
            (_Float16)o;
      }
    }
  }
}

// ---------------- kernel 3: proj + window reverse + LN + residual ----------------
__global__ void proj_kernel(const _Float16* __restrict__ attnout,
                            const _Float16* __restrict__ wp,
                            const float* __restrict__ proj_b,
                            const float* __restrict__ x,
                            const float* __restrict__ g1,
                            const float* __restrict__ b1,
                            float* __restrict__ x1,
                            _Float16* __restrict__ x1h) {
  __shared__ __attribute__((aligned(64))) _Float16 sA[16 * C_DIM];
  __shared__ __attribute__((aligned(64))) float sOut[16 * C_DIM];
  __shared__ int sL[16];
  const int tile = blockIdx.x, tid = threadIdx.x;
  const int mr = tid & 15, hi = tid >> 4;

  for (int i = tid; i < 16 * C_DIM; i += 32)
    sA[i] = attnout[(size_t)(tile * 16) * C_DIM + i];
  if (tid < 16) {
    int g = tile * 16 + tid;
    int b_ = g / NTOK, n = g - b_ * NTOK;
    sL[tid] = win_to_flat(b_, n);
  }
  __syncthreads();

  v16h afr[3];
  for (int kc = 0; kc < 3; ++kc)
    afr[kc] = load_afrag(sA + mr * C_DIM + kc * 32 + hi * 16);

  for (int nt = 0; nt < 6; ++nt) {
    v8f acc = {};
    for (int kc = 0; kc < 3; ++kc) {
      v16h b = *(const v16h*)(wp + (nt * 16 + mr) * C_DIM + kc * 32 + hi * 16);
      acc = wmma16(afr[kc], b, acc);
    }
    float bias = proj_b[nt * 16 + mr];
    for (int r = 0; r < 8; ++r)
      sOut[(r + hi * 8) * C_DIM + nt * 16 + mr] = acc[r] + bias;
  }
  __syncthreads();

  // per-row LayerNorm + residual scatter (2 lanes / row)
  const int row = mr;
  float s1 = 0.0f, s2 = 0.0f;
  for (int c = hi * 48; c < hi * 48 + 48; ++c) {
    float v = sOut[row * C_DIM + c];
    s1 += v; s2 += v * v;
  }
  s1 += __shfl_xor(s1, 16);
  s2 += __shfl_xor(s2, 16);
  float mean = s1 / C_DIM;
  float var  = s2 / C_DIM - mean * mean;
  float inv  = rsqrtf(var + 1e-5f);
  const int l = sL[row];
  for (int c = hi * 48; c < hi * 48 + 48; ++c) {
    float v = (sOut[row * C_DIM + c] - mean) * inv * g1[c] + b1[c];
    float o = x[(size_t)l * C_DIM + c] + v;
    x1[(size_t)l * C_DIM + c]  = o;
    x1h[(size_t)l * C_DIM + c] = (_Float16)o;
  }
}

// ---------------- kernel 4: fc1 + exact GELU ----------------
__global__ void fc1_kernel(const _Float16* __restrict__ x1h,
                           const _Float16* __restrict__ w1,
                           const float* __restrict__ fc1_b,
                           _Float16* __restrict__ hh) {
  __shared__ __attribute__((aligned(64))) _Float16 sA[16 * C_DIM];
  const int tile = blockIdx.x, tid = threadIdx.x;
  const int mr = tid & 15, hi = tid >> 4;

  for (int i = tid; i < 16 * C_DIM; i += 32)
    sA[i] = x1h[(size_t)(tile * 16) * C_DIM + i];
  __syncthreads();

  v16h afr[3];
  for (int kc = 0; kc < 3; ++kc)
    afr[kc] = load_afrag(sA + mr * C_DIM + kc * 32 + hi * 16);

  for (int nt = 0; nt < 24; ++nt) {
    v8f acc = {};
    for (int kc = 0; kc < 3; ++kc) {
      v16h b = *(const v16h*)(w1 + (nt * 16 + mr) * C_DIM + kc * 32 + hi * 16);
      acc = wmma16(afr[kc], b, acc);
    }
    int col = nt * 16 + mr;
    float bias = fc1_b[col];
    for (int r = 0; r < 8; ++r) {
      int l = tile * 16 + r + hi * 8;
      float v = acc[r] + bias;
      float ge = 0.5f * v * (1.0f + erff(v * 0.70710678118654752f));
      hh[(size_t)l * MLPH + col] = (_Float16)ge;
    }
  }
}

// ---------------- kernel 5: fc2 + LN + residual -> out ----------------
__global__ void fc2_kernel(const _Float16* __restrict__ hh,
                           const _Float16* __restrict__ w2,
                           const float* __restrict__ fc2_b,
                           const float* __restrict__ x1,
                           const float* __restrict__ g2,
                           const float* __restrict__ b2,
                           float* __restrict__ out) {
  __shared__ __attribute__((aligned(64))) _Float16 sA[16 * MLPH];
  __shared__ __attribute__((aligned(64))) float sOut[16 * C_DIM];
  const int tile = blockIdx.x, tid = threadIdx.x;
  const int mr = tid & 15, hi = tid >> 4;

  for (int i = tid; i < 16 * MLPH; i += 32)
    sA[i] = hh[(size_t)(tile * 16) * MLPH + i];
  __syncthreads();

  for (int nt = 0; nt < 6; ++nt) {
    v8f acc = {};
    for (int kc = 0; kc < 12; ++kc) {
      v16h a = load_afrag(sA + mr * MLPH + kc * 32 + hi * 16);
      v16h b = *(const v16h*)(w2 + (nt * 16 + mr) * MLPH + kc * 32 + hi * 16);
      acc = wmma16(a, b, acc);
    }
    float bias = fc2_b[nt * 16 + mr];
    for (int r = 0; r < 8; ++r)
      sOut[(r + hi * 8) * C_DIM + nt * 16 + mr] = acc[r] + bias;
  }
  __syncthreads();

  const int row = mr;
  float s1 = 0.0f, s2 = 0.0f;
  for (int c = hi * 48; c < hi * 48 + 48; ++c) {
    float v = sOut[row * C_DIM + c];
    s1 += v; s2 += v * v;
  }
  s1 += __shfl_xor(s1, 16);
  s2 += __shfl_xor(s2, 16);
  float mean = s1 / C_DIM;
  float var  = s2 / C_DIM - mean * mean;
  float inv  = rsqrtf(var + 1e-5f);
  const int l = tile * 16 + row;
  for (int c = hi * 48; c < hi * 48 + 48; ++c) {
    float v = (sOut[row * C_DIM + c] - mean) * inv * g2[c] + b2[c];
    out[(size_t)l * C_DIM + c] = x1[(size_t)l * C_DIM + c] + v;
  }
}

// ---------------- launch ----------------
extern "C" void kernel_launch(void* const* d_in, const int* in_sizes, int n_in,
                              void* d_out, int out_size, void* d_ws, size_t ws_size,
                              hipStream_t stream) {
  (void)in_sizes; (void)n_in; (void)out_size; (void)ws_size;
  const float* x        = (const float*)d_in[0];
  const float* mask     = (const float*)d_in[1];
  const float* qkv_w    = (const float*)d_in[2];
  const float* qkv_b    = (const float*)d_in[3];
  const float* proj_w   = (const float*)d_in[4];
  const float* proj_b   = (const float*)d_in[5];
  const float* lscale   = (const float*)d_in[6];
  const float* relTable = (const float*)d_in[7];
  const float* n1g      = (const float*)d_in[8];
  const float* n1b      = (const float*)d_in[9];
  const float* n2g      = (const float*)d_in[10];
  const float* n2b      = (const float*)d_in[11];
  const float* fc1_w    = (const float*)d_in[12];
  const float* fc1_b    = (const float*)d_in[13];
  const float* fc2_w    = (const float*)d_in[14];
  const float* fc2_b    = (const float*)d_in[15];
  float* out = (float*)d_out;

  char* ws = (char*)d_ws;
  size_t off = 0;
  _Float16* qkvh  = (_Float16*)(ws + off); off += (size_t)3 * C_DIM * C_DIM * 2;   // 55296
  _Float16* projh = (_Float16*)(ws + off); off += (size_t)C_DIM * C_DIM * 2;       // 18432
  _Float16* fc1h  = (_Float16*)(ws + off); off += (size_t)MLPH * C_DIM * 2;        // 73728
  _Float16* fc2h  = (_Float16*)(ws + off); off += (size_t)C_DIM * MLPH * 2;        // 73728
  float*    scale = (float*)(ws + off);    off += 256;
  const size_t qsz = (size_t)NWIN * 3 * NPAD * HD * 2;                             // 17.3 MB
  _Float16* qh = (_Float16*)(ws + off); off += qsz;
  _Float16* kh = (_Float16*)(ws + off); off += qsz;
  _Float16* vh = (_Float16*)(ws + off); off += qsz;
  _Float16* attnouth = (_Float16*)(ws + off); off += (size_t)LTOK * C_DIM * 2;     // 16.9 MB
  float*    x1  = (float*)(ws + off);       off += (size_t)LTOK * C_DIM * 4;       // 33.7 MB
  _Float16* x1h = (_Float16*)(ws + off);    off += (size_t)LTOK * C_DIM * 2;       // 16.9 MB
  _Float16* hh  = (_Float16*)(ws + off);    off += (size_t)LTOK * MLPH * 2;        // 67.4 MB

  prep_kernel<<<432, 256, 0, stream>>>(qkv_w, proj_w, fc1_w, fc2_w, lscale,
                                       qkvh, projh, fc1h, fc2h, scale);
  qkv_kernel<<<MTILES, 32, 0, stream>>>(x, qkvh, qkv_b, qh, kh, vh);
  attn_kernel<<<dim3(22, HEADS, NWIN), 32, 0, stream>>>(qh, kh, vh, scale,
                                                        relTable, mask, attnouth);
  proj_kernel<<<MTILES, 32, 0, stream>>>(attnouth, projh, proj_b, x, n1g, n1b,
                                         x1, x1h);
  fc1_kernel<<<MTILES, 32, 0, stream>>>(x1h, fc1h, fc1_b, hh);
  fc2_kernel<<<MTILES, 32, 0, stream>>>(hh, fc2h, fc2_b, x1, n2g, n2b, out);
}
